// TransformerLayer_5763846111739
// MI455X (gfx1250) — matrix operand order repaired and checked by reading, not compiled, not run on hardware
//
#include <hip/hip_runtime.h>

// ---------------- types ----------------
typedef __bf16 bf16;
typedef __attribute__((ext_vector_type(16))) __bf16 v16bf;
typedef __attribute__((ext_vector_type(8)))  float  v8f;
typedef __attribute__((ext_vector_type(4)))  __bf16 bf16x4;
typedef __attribute__((ext_vector_type(8)))  __bf16 bf16x8;

#define DEV static __device__ __forceinline__

// ---------------- problem constants ----------------
static constexpr int E_DIM   = 1024;
static constexpr int H_NUM   = 16;
static constexpr int D_HEAD  = 64;
static constexpr int N_SEQ   = 1024;
static constexpr int B_BATCH = 4;
static constexpr int MAXL    = 512;                 // rel-pos window
static constexpr int ROWS    = B_BATCH * N_SEQ;     // 4096
#define NEG_INF (-__builtin_inff())

DEV v8f zero8() {
  v8f z;
#pragma unroll
  for (int i = 0; i < 8; ++i) z[i] = 0.f;
  return z;
}

// 16-bit A-fragment (16x32): lane holds K = 16*kh+{0..15}
// -> two contiguous 16B chunks at p and p+8 (p already includes the 16*kh shift).
DEV v16bf ld_frag_a(const bf16* p) {
  bf16x8 lo = *(const bf16x8*)p;
  bf16x8 hi = *(const bf16x8*)(p + 8);
  return __builtin_shufflevector(lo, hi, 0, 1, 2, 3, 4, 5, 6, 7,
                                 8, 9, 10, 11, 12, 13, 14, 15);
}
// 16-bit B-fragment (Kx16): lane holds 16 consecutive K values -> one 32B chunk.
DEV v16bf ld_frag_b(const bf16* p) {
  bf16x8 lo = *(const bf16x8*)p;
  bf16x8 hi = *(const bf16x8*)(p + 8);
  return __builtin_shufflevector(lo, hi, 0, 1, 2, 3, 4, 5, 6, 7,
                                 8, 9, 10, 11, 12, 13, 14, 15);
}

// ---------------- f32 -> bf16 convert (weights) ----------------
__global__ void cvt_kernel(const float* __restrict__ in, bf16* __restrict__ out, int n) {
  int i = blockIdx.x * blockDim.x + threadIdx.x;
  int stride = gridDim.x * blockDim.x;
  for (; i < n; i += stride) out[i] = (bf16)in[i];
}

// ---------------- LayerNorm (one 256-thread block per row of 1024) ----------------
__global__ void __launch_bounds__(256)
ln_kernel(const float* __restrict__ x, const float* __restrict__ g,
          const float* __restrict__ bb, bf16* __restrict__ y) {
  const int row = blockIdx.x;
  const int tid = threadIdx.x;
  const float4 xv = ((const float4*)(x + (size_t)row * E_DIM))[tid];
  float s = xv.x + xv.y + xv.z + xv.w;
  float q = xv.x * xv.x + xv.y * xv.y + xv.z * xv.z + xv.w * xv.w;
#pragma unroll
  for (int off = 16; off; off >>= 1) {
    s += __shfl_xor(s, off, 32);
    q += __shfl_xor(q, off, 32);
  }
  __shared__ float ss[8], sq[8];
  const int wv = tid >> 5, lane = tid & 31;
  if (lane == 0) { ss[wv] = s; sq[wv] = q; }
  __syncthreads();
  if (tid == 0) {
    float ts = 0.f, tq = 0.f;
#pragma unroll
    for (int i = 0; i < 8; ++i) { ts += ss[i]; tq += sq[i]; }
    float mu  = ts * (1.f / E_DIM);
    float var = tq * (1.f / E_DIM) - mu * mu;
    ss[0] = mu; sq[0] = rsqrtf(var + 1e-5f);
  }
  __syncthreads();
  const float mu = ss[0], rstd = sq[0];
  const float4 gv = ((const float4*)g)[tid];
  const float4 bv = ((const float4*)bb)[tid];
  bf16x4 o;
  o[0] = (bf16)((xv.x - mu) * rstd * gv.x + bv.x);
  o[1] = (bf16)((xv.y - mu) * rstd * gv.y + bv.y);
  o[2] = (bf16)((xv.z - mu) * rstd * gv.z + bv.z);
  o[3] = (bf16)((xv.w - mu) * rstd * gv.w + bv.w);
  ((bf16x4*)(y + (size_t)row * E_DIM))[tid] = o;
}

// ---------------- WMMA GEMM:  out[M,Nt] = A[M,K] @ W[Nt,K]^T + bias (+ epilogue) --------
// Block = 256 threads (8 waves). Wave w: rows m0 = by*128 + w*16, cols n0 = bx*64.
// B tile (32k x 64n) staged in LDS (shared by all 8 waves).
// MODE 0: bf16 qkv out; V part (cols>=2048) additionally written transposed [B,H,D,N]
// MODE 1: f32 out + residual | MODE 2: bf16 out + ReLU
template <int MODE>
__global__ void __launch_bounds__(256)
gemm_kernel(const bf16* __restrict__ A, const bf16* __restrict__ W,
            const float* __restrict__ bias, const float* __restrict__ res,
            void* __restrict__ outp, bf16* __restrict__ vt,
            int M, int Nt, int K) {
  const int tid  = threadIdx.x;
  const int wv   = tid >> 5, lane = tid & 31;
  const int ln   = lane & 15, kh = lane >> 4;
  const int n0   = blockIdx.x * 64;
  const int m0   = blockIdx.y * 128 + wv * 16;

  __shared__ bf16 ldsB[64][40];   // 80B rows: every b128 access 16B-aligned

  v8f acc[4];
#pragma unroll
  for (int t = 0; t < 4; ++t) acc[t] = zero8();

  const bf16* Arow = A + (size_t)(m0 + ln) * K;
  const int nldr = tid >> 2;           // 0..63: which n this thread stages
  const int kseg = (tid & 3) * 8;      // 8 consecutive k (16 bytes)
  const bf16* Wld = W + (size_t)(n0 + nldr) * K + kseg;

  for (int kb = 0; kb < K; kb += 32) {
    bf16x8 w8 = *(const bf16x8*)(Wld + kb);
    if (kb + 32 < K) __builtin_prefetch(Wld + kb + 32, 0, 3);  // global_prefetch
    __syncthreads();                       // previous iteration's reads done
    *(bf16x8*)(&ldsB[nldr][kseg]) = w8;
    __syncthreads();

    const v16bf a = ld_frag_a(Arow + kb + 16 * kh);
#pragma unroll
    for (int t = 0; t < 4; ++t) {
      const v16bf bfrg = ld_frag_b(&ldsB[t * 16 + ln][16 * kh]);
      acc[t] = __builtin_amdgcn_wmma_f32_16x16x32_bf16(
          false, a, false, bfrg, (short)0, acc[t], false, false);
    }
  }

#pragma unroll
  for (int t = 0; t < 4; ++t) {
#pragma unroll
    for (int r = 0; r < 8; ++r) {
      const int row = m0 + r + 8 * kh;
      const int col = n0 + t * 16 + ln;
      const size_t idx = (size_t)row * Nt + col;
      float c = acc[t][r] + bias[col];
      if (MODE == 0) {
        ((bf16*)outp)[idx] = (bf16)c;
        if (col >= 2 * E_DIM) {            // V: also store transposed [B,H,D,N]
          const int b = row >> 10, i = row & (N_SEQ - 1);
          const int cv = col - 2 * E_DIM;
          const int h = cv >> 6, d = cv & (D_HEAD - 1);
          vt[(size_t)(((b * H_NUM + h) * D_HEAD) + d) * N_SEQ + i] = (bf16)c;
        }
      } else if (MODE == 1) {
        ((float*)outp)[idx] = c + res[idx];
      } else {
        ((bf16*)outp)[idx] = (bf16)fmaxf(c, 0.f);
      }
    }
  }
}

// ---------------- banded flash-attention with rel-pos bias ----------------
// One wave per 16-query tile of one (b,h). Q resident as two 16x32 A-frags.
// Keys streamed 32 at a time within window (0 <= i-j < 512). QK^T and P@V via WMMA.
// V read from the transposed [B,H,D,N] buffer -> contiguous B-fragments.
__global__ void __launch_bounds__(32)
attn_kernel(const bf16* __restrict__ qkv, const bf16* __restrict__ vt,
            const float* __restrict__ rel_pos, bf16* __restrict__ obuf) {
  const int lane = threadIdx.x & 31;
  const int ln = lane & 15, kh = lane >> 4;
  const int qt = blockIdx.x, h = blockIdx.y, b = blockIdx.z;
  const int i0 = qt * 16;
  const float SCALE = 0.125f;   // 1/sqrt(64)

  __shared__ bf16 pt[16][40];   // P bounce, 80B rows (all b128 aligned)

  // Q fragments (16 rows x 64 d, two K=32 steps)
  v16bf qa[2];
  {
    const size_t qrow = (size_t)(b * N_SEQ + i0 + ln) * (3 * E_DIM) + h * D_HEAD;
    qa[0] = ld_frag_a(qkv + qrow + 16 * kh);
    qa[1] = ld_frag_a(qkv + qrow + 32 + 16 * kh);
  }

  v8f oacc[4];
#pragma unroll
  for (int t = 0; t < 4; ++t) oacc[t] = zero8();
  float mrow[8], lrow[8];
#pragma unroll
  for (int r = 0; r < 8; ++r) { mrow[r] = NEG_INF; lrow[r] = 0.f; }

  int jb_lo = i0 - (MAXL - 1); if (jb_lo < 0) jb_lo = 0; jb_lo &= ~31;
  const int jhi = i0 + 15;

  for (int jb = jb_lo; jb <= jhi; jb += 32) {
    // ---- scores: two 16x16 tiles (keys jb..jb+15, jb+16..jb+31) ----
    v8f s2[2];
#pragma unroll
    for (int t = 0; t < 2; ++t) {
      const int j0 = jb + t * 16;
      const int jc = j0 + ln < N_SEQ ? j0 + ln : N_SEQ - 1;  // clamp; bias masks
      const size_t krow = (size_t)(b * N_SEQ + jc) * (3 * E_DIM) + E_DIM + h * D_HEAD;
      v8f s = zero8();
      s = __builtin_amdgcn_wmma_f32_16x16x32_bf16(
          false, qa[0], false, ld_frag_b(qkv + krow + 16 * kh), (short)0, s, false, false);
      s = __builtin_amdgcn_wmma_f32_16x16x32_bf16(
          false, qa[1], false, ld_frag_b(qkv + krow + 32 + 16 * kh), (short)0, s, false, false);
      s2[t] = s;
    }

    // ---- bias + online softmax (rows live at r + 8*kh, cols at lane&15) ----
#pragma unroll
    for (int r = 0; r < 8; ++r) {
      const int i = i0 + r + 8 * kh;
      int j = jb + ln, rel = i - j;
      float bia = (rel >= 0 && rel < MAXL) ? rel_pos[h * MAXL + rel] : NEG_INF;
      float v0 = s2[0][r] * SCALE + bia;
      j = jb + 16 + ln; rel = i - j;
      bia = (rel >= 0 && rel < MAXL) ? rel_pos[h * MAXL + rel] : NEG_INF;
      float v1 = s2[1][r] * SCALE + bia;

      float rmax = fmaxf(v0, v1);
#pragma unroll
      for (int off = 8; off; off >>= 1) rmax = fmaxf(rmax, __shfl_xor(rmax, off, 32));
      const float mnew = fmaxf(mrow[r], rmax);
      float alpha, p0, p1;
      if (mnew == NEG_INF) { alpha = 1.f; p0 = 0.f; p1 = 0.f; }
      else {
        alpha = __expf(mrow[r] - mnew);
        p0 = __expf(v0 - mnew);
        p1 = __expf(v1 - mnew);
      }
      float rs = p0 + p1;
#pragma unroll
      for (int off = 8; off; off >>= 1) rs += __shfl_xor(rs, off, 32);
      lrow[r] = lrow[r] * alpha + rs;
      mrow[r] = mnew;
#pragma unroll
      for (int t = 0; t < 4; ++t) oacc[t][r] *= alpha;
      pt[r + 8 * kh][ln]      = (bf16)p0;   // stage P (single wave: DS in-order)
      pt[r + 8 * kh][16 + ln] = (bf16)p1;
    }

    // ---- P back as A-fragment ----
    const v16bf pa = ld_frag_a(&pt[ln][16 * kh]);

    // ---- O += P @ V : four 16-col tiles over D=64, contiguous V-frags ----
    int jbase = jb + 16 * kh;
    if (jbase > N_SEQ - 16) jbase = N_SEQ - 16;   // clamp (clamped cols hit P==0)
#pragma unroll
    for (int t = 0; t < 4; ++t) {
      const int d = t * 16 + ln;
      const v16bf vf =
          ld_frag_b(vt + (size_t)((b * H_NUM + h) * D_HEAD + d) * N_SEQ + jbase);
      oacc[t] = __builtin_amdgcn_wmma_f32_16x16x32_bf16(
          false, pa, false, vf, (short)0, oacc[t], false, false);
    }
  }

  // ---- normalize and write o (merged back into [B,N,E]) ----
#pragma unroll
  for (int r = 0; r < 8; ++r) {
    const int i = i0 + r + 8 * kh;
    const float inv = lrow[r] > 0.f ? 1.f / lrow[r] : 0.f;
    const size_t orow = (size_t)(b * N_SEQ + i) * E_DIM + h * D_HEAD;
#pragma unroll
    for (int t = 0; t < 4; ++t)
      obuf[orow + t * 16 + ln] = (bf16)(oacc[t][r] * inv);
  }
}

// ---------------- launch ----------------
extern "C" void kernel_launch(void* const* d_in, const int* in_sizes, int n_in,
                              void* d_out, int out_size, void* d_ws, size_t ws_size,
                              hipStream_t stream) {
  const float* x         = (const float*)d_in[0];
  const float* rel_pos   = (const float*)d_in[1];
  const float* in_proj_w = (const float*)d_in[2];
  const float* in_proj_b = (const float*)d_in[3];
  const float* out_w     = (const float*)d_in[4];
  const float* out_b     = (const float*)d_in[5];
  const float* w1        = (const float*)d_in[6];
  const float* b1        = (const float*)d_in[7];
  const float* w2        = (const float*)d_in[8];
  const float* b2        = (const float*)d_in[9];
  const float* ln1_g     = (const float*)d_in[10];
  const float* ln1_b     = (const float*)d_in[11];
  const float* ln2_g     = (const float*)d_in[12];
  const float* ln2_b     = (const float*)d_in[13];
  float* out = (float*)d_out;

  // workspace layout (bytes)
  char* ws = (char*)d_ws;
  size_t off = 0;
  auto take = [&](size_t bytes) { char* p = ws + off; off += (bytes + 255) & ~(size_t)255; return p; };
  bf16*  wq   = (bf16*)take((size_t)3 * E_DIM * E_DIM * 2);       // in_proj bf16
  bf16*  wo   = (bf16*)take((size_t)E_DIM * E_DIM * 2);
  bf16*  w1b  = (bf16*)take((size_t)4 * E_DIM * E_DIM * 2);
  bf16*  w2b  = (bf16*)take((size_t)4 * E_DIM * E_DIM * 2);
  bf16*  xn   = (bf16*)take((size_t)ROWS * E_DIM * 2);
  bf16*  qkv  = (bf16*)take((size_t)ROWS * 3 * E_DIM * 2);
  bf16*  vtb  = (bf16*)take((size_t)ROWS * E_DIM * 2);            // V transposed [B,H,D,N]
  bf16*  obuf = (bf16*)take((size_t)ROWS * E_DIM * 2);
  float* x1   = (float*)take((size_t)ROWS * E_DIM * 4);
  bf16*  xm   = (bf16*)take((size_t)ROWS * E_DIM * 2);
  bf16*  hbuf = (bf16*)take((size_t)ROWS * 4 * E_DIM * 2);

  // 1) weights -> bf16
  cvt_kernel<<<2048, 256, 0, stream>>>(in_proj_w, wq, 3 * E_DIM * E_DIM);
  cvt_kernel<<<2048, 256, 0, stream>>>(out_w,     wo, E_DIM * E_DIM);
  cvt_kernel<<<2048, 256, 0, stream>>>(w1,        w1b, 4 * E_DIM * E_DIM);
  cvt_kernel<<<2048, 256, 0, stream>>>(w2,        w2b, 4 * E_DIM * E_DIM);

  // 2) LN1
  ln_kernel<<<ROWS, 256, 0, stream>>>(x, ln1_g, ln1_b, xn);

  // 3) qkv = xn @ in_proj_w^T + b   -> bf16 (+ transposed V)
  gemm_kernel<0><<<dim3(3 * E_DIM / 64, ROWS / 128), 256, 0, stream>>>(
      xn, wq, in_proj_b, nullptr, qkv, vtb, ROWS, 3 * E_DIM, E_DIM);

  // 4) banded flash attention
  attn_kernel<<<dim3(N_SEQ / 16, H_NUM, B_BATCH), 32, 0, stream>>>(qkv, vtb, rel_pos, obuf);

  // 5) x1 = x + o @ out_w^T + out_b   -> f32
  gemm_kernel<1><<<dim3(E_DIM / 64, ROWS / 128), 256, 0, stream>>>(
      obuf, wo, out_b, x, x1, nullptr, ROWS, E_DIM, E_DIM);

  // 6) LN2, FFN
  ln_kernel<<<ROWS, 256, 0, stream>>>(x1, ln2_g, ln2_b, xm);
  gemm_kernel<2><<<dim3(4 * E_DIM / 64, ROWS / 128), 256, 0, stream>>>(
      xm, w1b, b1, nullptr, hbuf, nullptr, ROWS, 4 * E_DIM, E_DIM);
  gemm_kernel<1><<<dim3(E_DIM / 64, ROWS / 128), 256, 0, stream>>>(
      hbuf, w2b, b2, x1, out, nullptr, ROWS, E_DIM, 4 * E_DIM);

  (void)in_sizes; (void)n_in; (void)out_size; (void)ws_size;
}
